// Multi_DefectModel_22986664968805
// MI455X (gfx1250) — matrix operand order repaired and checked by reading, not compiled, not run on hardware
//
#include <hip/hip_runtime.h>
#include <hip/hip_bf16.h>
#include <math.h>
#include <stdint.h>

#define NNODES 16000
#define NEDGES 96000
#define NB     128
#define EMBD   768
#define HFD    512
#define NHEAD  4
#define NFSW   1024
#define NCLS   2
#define GATD   (NHEAD * HFD)   // 2048
#define KC     128             // K-chunk staged in LDS (divides 512/768/1024/2048)

typedef __attribute__((ext_vector_type(16))) __bf16 v16bf;
typedef __attribute__((ext_vector_type(8)))  float  v8f;

struct __align__(16) U4 { unsigned x, y, z, w; };
union Frag { v16bf v; U4 q[2]; };

// ---------- helpers ----------
__device__ __forceinline__ unsigned short f2bf_rne(float f) {
  unsigned u = __float_as_uint(f);
  unsigned r = 0x7fffu + ((u >> 16) & 1u);
  return (unsigned short)((u + r) >> 16);
}
__device__ __forceinline__ float bf2f(unsigned short h) {
  return __uint_as_float(((unsigned)h) << 16);
}
__device__ __forceinline__ unsigned fflip(float f) {      // monotonic float->uint
  unsigned u = __float_as_uint(f);
  return (u & 0x80000000u) ? ~u : (u | 0x80000000u);
}
__device__ __forceinline__ float funflip(unsigned u) {
  return __uint_as_float((u & 0x80000000u) ? (u & 0x7fffffffu) : ~u);
}
#define FLIP_NEGINF 0x007fffffu

// gfx1250 async copy: global -> LDS, 16 bytes per lane, tracked by ASYNCcnt
__device__ __forceinline__ void async_g2lds_b128(unsigned lds_byte_addr, const void* gaddr) {
  asm volatile("global_load_async_to_lds_b128 %0, %1, off"
               :: "v"(lds_byte_addr), "v"(gaddr) : "memory");
}
__device__ __forceinline__ void wait_asynccnt0() {
  asm volatile("s_wait_asynccnt 0x0" ::: "memory");
}

// ---------- elementwise: fp32 -> (hi,lo) bf16 planes, optional eval-BN fold ----------
__global__ void k_split(const float* __restrict__ X, unsigned short* __restrict__ hi,
                        unsigned short* __restrict__ lo, long n, int cols,
                        const float* __restrict__ gamma, const float* __restrict__ beta,
                        float rsq) {
  for (long i = (long)blockIdx.x * blockDim.x + threadIdx.x; i < n;
       i += (long)gridDim.x * blockDim.x) {
    float x = X[i];
    if (gamma) { int c = (int)(i % cols); x = x * (gamma[c] * rsq) + beta[c]; }
    unsigned short h = f2bf_rne(x);
    hi[i] = h;
    lo[i] = f2bf_rne(x - bf2f(h));
  }
}

// ---------- weights: fp32 [K x Nc] -> transposed bf16 planes [Nc x K] ----------
__global__ void k_splitT(const float* __restrict__ W, unsigned short* __restrict__ hiT,
                         unsigned short* __restrict__ loT, int K, int Nc) {
  long n = (long)K * Nc;
  for (long i = (long)blockIdx.x * blockDim.x + threadIdx.x; i < n;
       i += (long)gridDim.x * blockDim.x) {
    int k = (int)(i / Nc), c = (int)(i % Nc);
    float x = W[i];
    unsigned short h = f2bf_rne(x);
    long o = (long)c * K + k;
    hiT[o] = h;
    loT[o] = f2bf_rne(x - bf2f(h));
  }
}

// ---------- WMMA GEMM: C[M x Nc] = act(A[M x K] * W[K x Nc] + bias) ----------
// bf16x3: C ~= Ahi*Bhi + Ahi*Blo + Alo*Bhi  (fp32 accumulate)
// Block tile 32Mx64N: 8 waves as 2(M)x4(N) 16x16 tiles. B tile (64 cols, hi+lo)
// staged in LDS per K-chunk via global_load_async_to_lds_b128; A streams from global.
// grid = (Nc/64, M/32), block = 256.
__global__ void __launch_bounds__(256)
k_gemm(const unsigned short* __restrict__ Ahi, const unsigned short* __restrict__ Alo,
       const unsigned short* __restrict__ BThi, const unsigned short* __restrict__ BTlo,
       const float* __restrict__ bias, float* __restrict__ C, int K, int Nc, int act) {
  __shared__ unsigned short sB[2][64 * KC];          // 32 KB: [plane][col*KC + k]

  int lane = threadIdx.x & 31;
  int wave = threadIdx.x >> 5;
  int half = lane >> 4;        // 0: lanes 0-15, 1: lanes 16-31
  int r    = lane & 15;
  int mi   = wave >> 2;        // 0..1
  int ni   = wave & 3;         // 0..3
  int tileM  = blockIdx.y * 32 + mi * 16;
  int tileNb = blockIdx.x * 64;            // block's first output column
  int tileN  = tileNb + ni * 16;

  const unsigned short* arh = Ahi + (size_t)(tileM + r) * K + half * 8;
  const unsigned short* arl = Alo + (size_t)(tileM + r) * K + half * 8;
  // per-wave B base inside LDS (column = ni*16 + r of the block tile)
  const unsigned short* lbh = &sB[0][(ni * 16 + r) * KC + half * 8];
  const unsigned short* lbl = &sB[1][(ni * 16 + r) * KC + half * 8];
  unsigned sB_lds = (unsigned)(uintptr_t)&sB[0][0];  // LDS byte address of staging buffer

  const int PER_COL = (KC * 2) / 16;                 // 16-byte transfers per column chunk
  const int TOTAL   = 2 * 64 * PER_COL;              // 2048 transfers per chunk

  v8f acc = {};
  for (int kc = 0; kc < K; kc += KC) {
    // ---- stage B chunk (hi+lo, 64 cols x KC) into LDS via async DMA ----
    for (int i = threadIdx.x; i < TOTAL; i += 256) {
      int within = i & (PER_COL - 1);                // 0..15 (16B units along K)
      int colp   = i / PER_COL;                      // 0..127
      int plane  = colp >> 6;
      int col    = colp & 63;
      const unsigned short* g =
          (plane ? BTlo : BThi) + (size_t)(tileNb + col) * K + kc + within * 8;
      unsigned lds = sB_lds + (unsigned)(((plane * 64 + col) * KC + within * 8) * 2);
      async_g2lds_b128(lds, g);
    }
    wait_asynccnt0();
    __syncthreads();

    // ---- compute on the chunk ----
    for (int kl = 0; kl < KC; kl += 32) {
      int kk = kc + kl;
      Frag ah, al, bh, bl;
      // A 16x32 bf16 layout: lanes<16 hold K {0..7,16..23}, lanes>=16 hold {8..15,24..31}
      ah.q[0] = *(const U4*)(arh + kk);  ah.q[1] = *(const U4*)(arh + kk + 16);
      al.q[0] = *(const U4*)(arl + kk);  al.q[1] = *(const U4*)(arl + kk + 16);
      // B 32x16 bf16 layout from LDS: lanes<16 K {0..7,16..23}; lanes>=16 K {8..15,24..31}
      bh.q[0] = *(const U4*)(lbh + kl);  bh.q[1] = *(const U4*)(lbh + kl + 16);
      bl.q[0] = *(const U4*)(lbl + kl);  bl.q[1] = *(const U4*)(lbl + kl + 16);
      __builtin_prefetch(arh + kk + 32, 0, 3);
      acc = __builtin_amdgcn_wmma_f32_16x16x32_bf16(false, ah.v, false, bh.v, (short)0, acc, false, false);
      acc = __builtin_amdgcn_wmma_f32_16x16x32_bf16(false, ah.v, false, bl.v, (short)0, acc, false, false);
      acc = __builtin_amdgcn_wmma_f32_16x16x32_bf16(false, al.v, false, bh.v, (short)0, acc, false, false);
    }
    __syncthreads();   // protect LDS before next chunk's async writes
  }

  int col = tileN + r;
  float bv = bias ? bias[col] : 0.0f;
  float* crow = C + (size_t)(tileM + half * 8) * Nc + col;
#pragma unroll
  for (int i = 0; i < 8; ++i) {            // VGPR i holds rows i (lanes<16) / i+8 (lanes>=16)
    float v = acc[i] + bv;
    if (act) v = (v > 0.0f) ? v : expm1f(v);
    crow[(size_t)i * Nc] = v;
  }
}

// ---------- GAT attention pieces ----------
__global__ void k_attn_scores(const float* __restrict__ F, const float* __restrict__ al,
                              const float* __restrict__ ar, float* __restrict__ el,
                              float* __restrict__ er) {
  int n = blockIdx.x;
  int h = threadIdx.x >> 5;
  int lane = threadIdx.x & 31;
  const float* f = F + (size_t)n * GATD + h * HFD;
  const float* a = al + h * HFD;
  const float* b = ar + h * HFD;
  float sl = 0.f, sr = 0.f;
  for (int d = lane; d < HFD; d += 32) { float fv = f[d]; sl += fv * a[d]; sr += fv * b[d]; }
  for (int off = 16; off > 0; off >>= 1) {
    sl += __shfl_xor(sl, off, 32);
    sr += __shfl_xor(sr, off, 32);
  }
  if (lane == 0) { el[n * NHEAD + h] = sl; er[n * NHEAD + h] = sr; }
}

__global__ void k_init_attn(unsigned* __restrict__ emaxb, float* __restrict__ den, long n) {
  for (long i = (long)blockIdx.x * blockDim.x + threadIdx.x; i < n;
       i += (long)gridDim.x * blockDim.x) { emaxb[i] = FLIP_NEGINF; den[i] = 0.0f; }
}

__global__ void k_fill_bias(float* __restrict__ R, const float* __restrict__ bias,
                            long n, int cols) {
  for (long i = (long)blockIdx.x * blockDim.x + threadIdx.x; i < n;
       i += (long)gridDim.x * blockDim.x) R[i] = bias[i % cols];
}

__global__ void k_edge_max(const int* __restrict__ src, const int* __restrict__ dst,
                           const float* __restrict__ el, const float* __restrict__ er,
                           float* __restrict__ e, unsigned* __restrict__ emaxb) {
  long n = (long)NEDGES * NHEAD;
  for (long i = (long)blockIdx.x * blockDim.x + threadIdx.x; i < n;
       i += (long)gridDim.x * blockDim.x) {
    int eidx = (int)(i >> 2), h = (int)(i & 3);
    int s = src[eidx], d = dst[eidx];
    float v = el[s * NHEAD + h] + er[d * NHEAD + h];
    v = (v > 0.0f) ? v : 0.2f * v;                  // leaky_relu(0.2)
    e[i] = v;
    atomicMax(&emaxb[d * NHEAD + h], fflip(v));
  }
}

__global__ void k_emax_fin(const unsigned* __restrict__ emaxb, float* __restrict__ emaxf, long n) {
  for (long i = (long)blockIdx.x * blockDim.x + threadIdx.x; i < n;
       i += (long)gridDim.x * blockDim.x) {
    unsigned u = emaxb[i];
    emaxf[i] = (u == FLIP_NEGINF) ? 0.0f : funflip(u);   // where(isfinite, emax, 0)
  }
}

__global__ void k_edge_exp(const int* __restrict__ dst, const float* __restrict__ e,
                           const float* __restrict__ emaxf, float* __restrict__ ee,
                           float* __restrict__ den) {
  long n = (long)NEDGES * NHEAD;
  for (long i = (long)blockIdx.x * blockDim.x + threadIdx.x; i < n;
       i += (long)gridDim.x * blockDim.x) {
    int eidx = (int)(i >> 2), h = (int)(i & 3);
    int d = dst[eidx];
    float v = expf(e[i] - emaxf[d * NHEAD + h]);
    ee[i] = v;
    atomicAdd(&den[d * NHEAD + h], v);
  }
}

// one wave per (edge, head): rst[dst,h,:] += a * feat[src,h,:]
__global__ void __launch_bounds__(256)
k_agg(const int* __restrict__ src, const int* __restrict__ dst, const float* __restrict__ ee,
      const float* __restrict__ den, const float* __restrict__ F, float* __restrict__ rst) {
  long w = (long)blockIdx.x * 8 + (threadIdx.x >> 5);
  if (w >= (long)NEDGES * NHEAD) return;
  int lane = threadIdx.x & 31;
  int eidx = (int)(w >> 2), h = (int)(w & 3);
  int s = src[eidx], d = dst[eidx];
  float a = ee[w] / den[d * NHEAD + h];
  const float* fs = F + (size_t)s * GATD + h * HFD;
  float* rd = rst + (size_t)d * GATD + h * HFD;
  for (int k = lane; k < HFD; k += 32) atomicAdd(&rd[k], a * fs[k]);
}

// ---------- pooling ----------
__global__ void k_zero(float* __restrict__ p, long n) {
  for (long i = (long)blockIdx.x * blockDim.x + threadIdx.x; i < n;
       i += (long)gridDim.x * blockDim.x) p[i] = 0.0f;
}
__global__ void k_pool(const float* __restrict__ H, const int* __restrict__ gids,
                       float* __restrict__ pooled) {
  long n = (long)NNODES * HFD;
  for (long i = (long)blockIdx.x * blockDim.x + threadIdx.x; i < n;
       i += (long)gridDim.x * blockDim.x) {
    int nd = (int)(i / HFD), c = (int)(i % HFD);
    atomicAdd(&pooled[(size_t)gids[nd] * HFD + c], H[i]);
  }
}
__global__ void k_cnt(const int* __restrict__ gids, float* __restrict__ cnt) {
  for (long i = (long)blockIdx.x * blockDim.x + threadIdx.x; i < NNODES;
       i += (long)gridDim.x * blockDim.x) atomicAdd(&cnt[gids[i]], 1.0f);
}
__global__ void k_mean(float* __restrict__ pooled, const float* __restrict__ cnt) {
  long n = (long)NB * HFD;
  for (long i = (long)blockIdx.x * blockDim.x + threadIdx.x; i < n;
       i += (long)gridDim.x * blockDim.x) pooled[i] /= fmaxf(cnt[i / HFD], 1.0f);
}

// ---------- final: out = bn(concat[x,hg,t]) @ Wfin + bfin ----------
__global__ void k_final(const float* __restrict__ x, const float* __restrict__ hg,
                        const float* __restrict__ t, const float* __restrict__ gfin,
                        const float* __restrict__ befin, const float* __restrict__ Wfin,
                        const float* __restrict__ bfin, float* __restrict__ out, float rsq) {
  int tt = threadIdx.x;            // 256 threads: 128 graphs x 2 classes
  int b = tt >> 1, cls = tt & 1;
  float acc = 0.0f;
  for (int j = 0; j < 3 * HFD; ++j) {
    float v = (j < HFD) ? x[(size_t)b * HFD + j]
            : (j < 2 * HFD) ? hg[(size_t)b * HFD + (j - HFD)]
                            : t[(size_t)b * HFD + (j - 2 * HFD)];
    v = v * (gfin[j] * rsq) + befin[j];
    acc += v * Wfin[(size_t)j * NCLS + cls];
  }
  out[(size_t)b * NCLS + cls] = acc + bfin[cls];
}

// ================= host orchestration =================
extern "C" void kernel_launch(void* const* d_in, const int* in_sizes, int n_in,
                              void* d_out, int out_size, void* d_ws, size_t ws_size,
                              hipStream_t stream) {
  (void)in_sizes; (void)n_in; (void)out_size; (void)ws_size;
  const float* node_emb = (const float*)d_in[0];
  // d_in[1] = func_emb: dead code (h_func never reaches the output) -> skipped
  const float* img_emb  = (const float*)d_in[2];
  const float* text_emb = (const float*)d_in[3];
  const int*   src      = (const int*)d_in[4];
  const int*   dst      = (const int*)d_in[5];
  const int*   gids     = (const int*)d_in[6];
  const float* W1   = (const float*)d_in[7];
  const float* al1  = (const float*)d_in[8];
  const float* ar1  = (const float*)d_in[9];
  const float* b1   = (const float*)d_in[10];
  const float* W2   = (const float*)d_in[11];
  const float* al2  = (const float*)d_in[12];
  const float* ar2  = (const float*)d_in[13];
  const float* b2   = (const float*)d_in[14];
  const float* Wfc  = (const float*)d_in[15];
  const float* bfc  = (const float*)d_in[16];
  // d_in[17] Wfo, d_in[18] bfo: dead branch
  const float* Wh   = (const float*)d_in[19];
  const float* bh   = (const float*)d_in[20];
  const float* g_text = (const float*)d_in[21];
  const float* be_text= (const float*)d_in[22];
  const float* Wtext  = (const float*)d_in[23];
  const float* btext  = (const float*)d_in[24];
  const float* g_swin = (const float*)d_in[25];
  const float* be_swin= (const float*)d_in[26];
  const float* Wswin  = (const float*)d_in[27];
  const float* bswin  = (const float*)d_in[28];
  const float* g_h    = (const float*)d_in[29];
  const float* be_h   = (const float*)d_in[30];
  const float* Whfc   = (const float*)d_in[31];
  const float* bhfc   = (const float*)d_in[32];
  const float* g_fin  = (const float*)d_in[33];
  const float* be_fin = (const float*)d_in[34];
  const float* Wfin   = (const float*)d_in[35];
  const float* bfin   = (const float*)d_in[36];
  float* out = (float*)d_out;

  const float rsq = 1.0f / sqrtf(1.0f + 1e-5f);

  // ---- scratch arena ----
  char* w = (char*)d_ws;
  size_t off = 0;
  auto give = [&](size_t bytes) { size_t r = off; off += (bytes + 255) & ~(size_t)255; return r; };
  const size_t SA = (size_t)NNODES * GATD * 2;      // bf16 activation plane (max)
  const size_t SW = (size_t)GATD * GATD * 2;        // bf16 weight plane (max)
  const size_t SF = (size_t)NNODES * GATD * 4;      // fp32 feature buffer (max)
  unsigned short* AHI = (unsigned short*)(w + give(SA));
  unsigned short* ALO = (unsigned short*)(w + give(SA));
  unsigned short* WHI = (unsigned short*)(w + give(SW));
  unsigned short* WLO = (unsigned short*)(w + give(SW));
  float* F    = (float*)(w + give(SF));
  float* R    = (float*)(w + give(SF));
  float* EL   = (float*)(w + give((size_t)NNODES * NHEAD * 4));
  float* ER   = (float*)(w + give((size_t)NNODES * NHEAD * 4));
  unsigned* EMAXB = (unsigned*)(w + give((size_t)NNODES * NHEAD * 4));
  float* EMAXF= (float*)(w + give((size_t)NNODES * NHEAD * 4));
  float* DEN  = (float*)(w + give((size_t)NNODES * NHEAD * 4));
  float* EBUF = (float*)(w + give((size_t)NEDGES * NHEAD * 4));
  float* EEBUF= (float*)(w + give((size_t)NEDGES * NHEAD * 4));
  float* POOL = (float*)(w + give((size_t)NB * HFD * 4));
  float* CNT  = (float*)(w + give((size_t)NB * 4));
  float* XB   = (float*)(w + give((size_t)NB * HFD * 4));
  float* TB   = (float*)(w + give((size_t)NB * HFD * 4));
  float* HG   = (float*)(w + give((size_t)NB * HFD * 4));

  auto gemm = [&](const float* bias, float* C, int M, int K, int Nc, int act) {
    k_gemm<<<dim3(Nc / 64, M / 32), 256, 0, stream>>>(AHI, ALO, WHI, WLO, bias, C, K, Nc, act);
  };
  auto gat_attn = [&](const float* Feat, float* Rst, const float* al, const float* ar,
                      const float* bias) {
    k_attn_scores<<<NNODES, NHEAD * 32, 0, stream>>>(Feat, al, ar, EL, ER);
    k_init_attn<<<250, 256, 0, stream>>>(EMAXB, DEN, (long)NNODES * NHEAD);
    k_fill_bias<<<4096, 256, 0, stream>>>(Rst, bias, (long)NNODES * GATD, GATD);
    k_edge_max<<<1500, 256, 0, stream>>>(src, dst, EL, ER, EBUF, EMAXB);
    k_emax_fin<<<250, 256, 0, stream>>>(EMAXB, EMAXF, (long)NNODES * NHEAD);
    k_edge_exp<<<1500, 256, 0, stream>>>(dst, EBUF, EMAXF, EEBUF, DEN);
    k_agg<<<(NEDGES * NHEAD) / 8, 256, 0, stream>>>(src, dst, EEBUF, DEN, Feat, Rst);
  };

  // ===== GAT layer 1: feat1 = node_emb @ W1; rst1 = edge-softmax-agg + b1 =====
  k_split<<<4096, 256, 0, stream>>>(node_emb, AHI, ALO, (long)NNODES * EMBD, 1, nullptr, nullptr, rsq);
  k_splitT<<<4096, 256, 0, stream>>>(W1, WHI, WLO, EMBD, GATD);
  gemm(nullptr, F, NNODES, EMBD, GATD, 0);
  gat_attn(F, R, al1, ar1, b1);

  // ===== GAT layer 2 =====
  k_split<<<4096, 256, 0, stream>>>(R, AHI, ALO, (long)NNODES * GATD, 1, nullptr, nullptr, rsq);
  k_splitT<<<4096, 256, 0, stream>>>(W2, WHI, WLO, GATD, GATD);
  gemm(nullptr, F, NNODES, GATD, GATD, 0);
  gat_attn(F, R, al2, ar2, b2);

  // ===== fc: h = elu(R @ Wfc + bfc)  -> F as [N, 512] =====
  k_split<<<4096, 256, 0, stream>>>(R, AHI, ALO, (long)NNODES * GATD, 1, nullptr, nullptr, rsq);
  k_splitT<<<4096, 256, 0, stream>>>(Wfc, WHI, WLO, GATD, HFD);
  gemm(bfc, F, NNODES, GATD, HFD, 1);

  // ===== 8 shared hidden layers on h =====
  for (int i = 0; i < 8; ++i) {
    k_split<<<4096, 256, 0, stream>>>(F, AHI, ALO, (long)NNODES * HFD, 1, nullptr, nullptr, rsq);
    k_splitT<<<2048, 256, 0, stream>>>(Wh + (size_t)i * HFD * HFD, WHI, WLO, HFD, HFD);
    gemm(bh + (size_t)i * HFD, F, NNODES, HFD, HFD, 1);
  }

  // ===== per-graph mean pooling =====
  k_zero<<<256, 256, 0, stream>>>(POOL, (long)NB * HFD);
  k_zero<<<1, 128, 0, stream>>>(CNT, NB);
  k_pool<<<4096, 256, 0, stream>>>(F, gids, POOL);
  k_cnt<<<64, 256, 0, stream>>>(gids, CNT);
  k_mean<<<256, 256, 0, stream>>>(POOL, CNT);

  // ===== hg = elu(bn(hmean) @ Whfc + bhfc) =====
  k_split<<<256, 256, 0, stream>>>(POOL, AHI, ALO, (long)NB * HFD, HFD, g_h, be_h, rsq);
  k_splitT<<<2048, 256, 0, stream>>>(Whfc, WHI, WLO, HFD, HFD);
  gemm(bhfc, HG, NB, HFD, HFD, 1);

  // ===== image branch: x = elu(bn(img) @ Wswin + bswin) =====
  k_split<<<512, 256, 0, stream>>>(img_emb, AHI, ALO, (long)NB * NFSW, NFSW, g_swin, be_swin, rsq);
  k_splitT<<<4096, 256, 0, stream>>>(Wswin, WHI, WLO, NFSW, HFD);
  gemm(bswin, XB, NB, NFSW, HFD, 1);

  // ===== text branch: t = elu(bn(text) @ Wtext + btext) =====
  k_split<<<512, 256, 0, stream>>>(text_emb, AHI, ALO, (long)NB * EMBD, EMBD, g_text, be_text, rsq);
  k_splitT<<<2048, 256, 0, stream>>>(Wtext, WHI, WLO, EMBD, HFD);
  gemm(btext, TB, NB, EMBD, HFD, 1);

  // ===== final classifier =====
  k_final<<<1, 256, 0, stream>>>(XB, HG, TB, g_fin, be_fin, Wfin, bfin, out, rsq);
}
